// DAU_ours_v5_29472065585825
// MI455X (gfx1250) — matrix operand order repaired and checked by reading, not compiled, not run on hardware
//
#include <hip/hip_runtime.h>
#include <hip/hip_bf16.h>

typedef __attribute__((ext_vector_type(16))) _Float16 v16h;
typedef __attribute__((ext_vector_type(8)))  _Float16 v8h;
typedef __attribute__((ext_vector_type(8)))  float    v8f;
typedef __attribute__((ext_vector_type(4))) unsigned int u32x4;
typedef __attribute__((ext_vector_type(8))) int          i32x8;
typedef __attribute__((ext_vector_type(4))) int          i32x4;

#define HH 128
#define WW 128
#define HWP (HH * WW)          // 16384 = 2^14
#define NB 8
#define NC 64
#define EELEM (8ull * 64 * 128 * 128)   // 8,388,608

#if defined(__has_builtin)
#if __has_builtin(__builtin_amdgcn_tensor_load_to_lds) && __has_builtin(__builtin_amdgcn_s_wait_tensorcnt)
#define HAS_TDM 1
#endif
#endif

__device__ __forceinline__ int iclamp(int v, int lo, int hi) {
    return v < lo ? lo : (v > hi ? hi : v);
}

union HU16 { v16h v; v8h h[2]; };

#ifdef HAS_TDM
// Issue one TDM tile load: 3D tile [nch][rows][wst] (x fastest) from a
// [C][H][W] f16 tensor into LDS at lds_addr.  D# built per ISA ch.8.
__device__ __forceinline__ void tdm_load_tile(unsigned long long gaddr,
                                              unsigned lds_addr,
                                              int wst, int rows, int nch) {
    u32x4 g0 = { 1u,                                   // count=1, user mode
                 lds_addr,                             // lds_addr [63:32]
                 (unsigned)gaddr,                      // global_addr low
                 (unsigned)((gaddr >> 32) & 0x01FFFFFFu) | 0x80000000u }; // type=2
    // group1: data_size=1 (2B); tensor_dim0/1 = 1<<20 (no OOB clip);
    // tile dims; dim0_stride=WW, dim1_stride=HWP
    i32x8 g1 = { 0x10000,                              // data_size=2B
                 0,                                    // atomic_addr=0, dim0 lo16=0
                 16,                                   // dim0 hi16=16 (1M), dim1 lo16=0
                 16 | (wst << 16),                     // dim1 hi16, tile_dim0
                 rows | (nch << 16),                   // tile_dim1, tile_dim2
                 WW,                                   // dim0_stride lo32
                 (int)(((unsigned)HWP & 0xFFFFu) << 16), // dim1_stride lo16 in [31:16]
                 0 };                                  // dim1_stride hi
    i32x4 g2 = { 1 << 20, 0, 0, 0 };                   // tensor_dim2 large
    i32x4 g3 = { 0, 0, 0, 0 };
#if __clang_major__ >= 23
    i32x8 gz = { 0, 0, 0, 0, 0, 0, 0, 0 };
    __builtin_amdgcn_tensor_load_to_lds(g0, g1, g2, g3, gz, 0);
#else
    __builtin_amdgcn_tensor_load_to_lds(g0, g1, g2, g3, 0);
#endif
}
#endif

// ---------------------------------------------------------------------------
// Implicit-GEMM conv via WMMA f16->f32, fully unrolled via templates.
//   Wp : packed f16 weights [tap][COUT][CIN]
//   X0 : f16 activations [B][Cin0][H][W]; X1: channels Cin0..CIN-1 (concat)
//   Yf/Yh : optional f32 / f16 outputs;  resid: optional f32 residual add
//   bias[co*bstride + boff] optional;  order: +bias -> relu -> +resid
//   Staging: TDM (tensor_load_to_lds) pulls the raw [c][r][w] tile into LDS
//   (wave 0, TENSORcnt); a transpose pass builds the [r][w][c] K-contiguous
//   tile (and applies the zero halo), so each WMMA B-fragment is two aligned
//   ds_load_b128.  Falls back to direct global staging without TDM support.
// ---------------------------------------------------------------------------
template<int KS, int CIN, int COUT, int NPIX>
__global__ __launch_bounds__(256)
void k_conv_wmma(const _Float16* __restrict__ Wp,
                 const _Float16* __restrict__ X0,
                 const _Float16* __restrict__ X1, int Cin0,
                 float* __restrict__ Yf,
                 _Float16* __restrict__ Yh,
                 const float* __restrict__ resid,
                 const float* __restrict__ bias, int bstride, int boff,
                 int relu_flag)
{
    constexpr int PAD  = (KS == 3) ? 1 : 0;
    constexpr int ROWS = (KS == 3) ? 3 : 1;
    constexpr int WST  = NPIX + 2 * PAD;
    constexpr int NT   = NPIX / 16;
    constexpr int MT   = COUT / 16;
    static_assert(MT * NT == 8, "8 waves per workgroup");

    const int b  = blockIdx.y;
    const int h  = blockIdx.x / (WW / NPIX);
    const int w0 = (blockIdx.x % (WW / NPIX)) * NPIX;

    __shared__ _Float16 sT[ROWS * WST * CIN];   // [r][wl][ci] transposed

    const int tid = threadIdx.y * 32 + threadIdx.x;
    constexpr int TOTAL = ROWS * WST * CIN;

    if (tid == 0) __builtin_prefetch(Wp, 0, 3);   // global_prefetch_b8

#ifdef HAS_TDM
    __shared__ _Float16 sRaw[ROWS * WST * CIN];  // [c][r][wl] raw TDM order
    if (threadIdx.y == 0) {
        // Tensor Data Mover: async DMA of the activation tile into LDS.
        const long long e0 =
            ((long long)(b * Cin0) << 14) + (h - PAD) * WW + (w0 - PAD);
        tdm_load_tile((unsigned long long)(uintptr_t)X0 + 2ll * e0,
                      (unsigned)(uintptr_t)&sRaw[0], WST, ROWS, Cin0);
        if (Cin0 < CIN) {
            const long long e1 =
                ((long long)(b * (CIN - Cin0)) << 14) + (h - PAD) * WW + (w0 - PAD);
            tdm_load_tile((unsigned long long)(uintptr_t)X1 + 2ll * e1,
                          (unsigned)(uintptr_t)&sRaw[Cin0 * ROWS * WST],
                          WST, ROWS, CIN - Cin0);
        }
        __builtin_amdgcn_s_wait_tensorcnt(0);
    }
    __syncthreads();
    // LDS->LDS transpose into K-contiguous layout + zero halo.
    for (int i = tid; i < TOTAL; i += 256) {
        int c   = i / (ROWS * WST);
        int rem = i - c * (ROWS * WST);
        int r   = rem / WST;
        int wl  = rem - r * WST;
        int hg  = h + r - PAD;
        int wg  = w0 + wl - PAD;
        bool inb = (hg >= 0 && hg < HH && wg >= 0 && wg < WW);
        sT[(r * WST + wl) * CIN + c] = inb ? sRaw[i] : (_Float16)0.f;
    }
#else
    // Fallback: direct global staging (coalesced reads, transposed writes).
    for (int i = tid; i < TOTAL; i += 256) {
        int c   = i / (ROWS * WST);
        int rem = i - c * (ROWS * WST);
        int r   = rem / WST;
        int wl  = rem - r * WST;
        int hg  = h + r - PAD;
        int wg  = w0 + wl - PAD;
        _Float16 v = (_Float16)0.f;
        if (hg >= 0 && hg < HH && wg >= 0 && wg < WW) {
            if (c < Cin0)
                v = X0[((size_t)(b * Cin0 + c) << 14) + hg * WW + wg];
            else
                v = X1[((size_t)(b * (CIN - Cin0) + (c - Cin0)) << 14) + hg * WW + wg];
        }
        sT[(r * WST + wl) * CIN + c] = v;
    }
#endif
    __syncthreads();

    const int wid   = threadIdx.y;
    const int mtile = wid / NT;
    const int ntile = wid - mtile * NT;
    const int lane  = threadIdx.x;
    const int half  = lane >> 4;
    const int l15   = lane & 15;
    const int co    = mtile * 16 + l15;    // A row for this lane
    const int ncol  = ntile * 16 + l15;    // B column for this lane

    v8f acc = {};

    #pragma unroll
    for (int tap = 0; tap < KS * KS; ++tap) {
        const int rr = tap / KS;
        const int ss = tap - rr * KS;
        const _Float16* wrow = Wp + (size_t)tap * COUT * CIN + (size_t)co * CIN;
        #pragma unroll
        for (int kc = 0; kc < CIN; kc += 32) {
            // A fragment: lane holds K = {8h..8h+7} U {8h+16..8h+23}
            HU16 A;
            A.h[0] = *reinterpret_cast<const v8h*>(wrow + kc + 8 * half);
            A.h[1] = *reinterpret_cast<const v8h*>(wrow + kc + 8 * half + 16);

            // B fragment: K = {8h..8h+7} U {8h+16..8h+23} at fixed column -> 2x ds_load_b128
            const int kbase = kc + 8 * half;
            const int wloc  = ncol + ss;
            const v8h* bp = reinterpret_cast<const v8h*>(
                sT + (rr * WST + wloc) * CIN + kbase);
            HU16 B;
            B.h[0] = bp[0];
            B.h[1] = bp[2];

            acc = __builtin_amdgcn_wmma_f32_16x16x32_f16(
                false, A.v, false, B.v, (short)0, acc, false, false);
        }
    }

    // Epilogue: C/D layout — VGPR r: M = r + 8*half, N = lane&15
    const int p0 = h * WW + w0;
    #pragma unroll
    for (int r8 = 0; r8 < 8; ++r8) {
        int coo = mtile * 16 + r8 + 8 * half;
        size_t gi = ((size_t)(b * COUT + coo) << 14) + p0 + ncol;
        float v = acc[r8];
        if (bias)      v += bias[coo * bstride + boff];
        if (relu_flag) v = v > 0.f ? v : 0.f;
        if (resid)     v += resid[gi];
        if (Yf) Yf[gi] = v;
        if (Yh) Yh[gi] = (_Float16)v;
    }
}

// ---------------------------------------------------------------------------
// Weight packing and elementwise glue kernels
// ---------------------------------------------------------------------------
__global__ void k_pack3(const float* __restrict__ w, _Float16* __restrict__ o,
                        int Cout, int Cin) {
    int i = blockIdx.x * 256 + threadIdx.x;
    int n = Cout * Cin * 9;
    if (i >= n) return;
    int tap = i / (Cout * Cin);
    int rem = i - tap * (Cout * Cin);
    int co  = rem / Cin;
    int ci  = rem - co * Cin;
    o[i] = (_Float16)w[((co * Cin + ci) * 9) + tap];
}

__global__ void k_packg2(const float* __restrict__ g2, _Float16* __restrict__ o) {
    int i = blockIdx.x * 256 + threadIdx.x;       // out [9][64][64]
    if (i >= 9 * 64 * 64) return;
    int tap = i >> 12;
    int rem = i & 4095;
    int c   = rem >> 6;
    int ci  = rem & 63;
    o[i] = (_Float16)g2[((c * 9 + tap) << 6) + ci];
}

__global__ void k_cast_h(const float* __restrict__ a, _Float16* __restrict__ o, size_t n) {
    size_t i = (size_t)blockIdx.x * 256 + threadIdx.x;
    if (i < n) o[i] = (_Float16)a[i];
}

__global__ void k_add2h(const float* __restrict__ a, const float* __restrict__ b,
                        _Float16* __restrict__ o, size_t n) {
    size_t i = (size_t)blockIdx.x * 256 + threadIdx.x;
    if (i < n) o[i] = (_Float16)(a[i] + b[i]);
}

__global__ void k_cmean(const float* __restrict__ x, float* __restrict__ w0) {
    int bc = blockIdx.x;                           // B*C blocks
    const float* p = x + ((size_t)bc << 14);
    __shared__ float red[256];
    float s = 0.f;
    for (int i = threadIdx.x; i < HWP; i += 256) s += p[i];
    red[threadIdx.x] = s;
    __syncthreads();
    for (int off = 128; off > 0; off >>= 1) {
        if (threadIdx.x < off) red[threadIdx.x] += red[threadIdx.x + off];
        __syncthreads();
    }
    if (threadIdx.x == 0) w0[bc] = red[0] * (1.f / (float)HWP);
}

__global__ void k_mlp(const float* __restrict__ w0,
                      const float* __restrict__ A1, const float* __restrict__ b1,
                      const float* __restrict__ A2, const float* __restrict__ b2,
                      float* __restrict__ w0s) {
    __shared__ float hid[8 * 128];
    for (int i = threadIdx.x; i < 8 * 128; i += blockDim.x) {
        int bb = i >> 7, j = i & 127;
        float s = b1[j];
        for (int c = 0; c < 64; ++c) s += A1[j * 64 + c] * w0[bb * 64 + c];
        hid[i] = s > 0.f ? s : 0.f;
    }
    __syncthreads();
    for (int i = threadIdx.x; i < 8 * 64; i += blockDim.x) {
        int bb = i >> 6, c = i & 63;
        float s = b2[c];
        for (int j = 0; j < 128; ++j) s += A2[c * 128 + j] * hid[bb * 128 + j];
        w0s[i] = 1.f / (1.f + expf(-s));
    }
}

__global__ void k_gate_ch(const float* __restrict__ x, const float* __restrict__ g,
                          float* __restrict__ o, size_t n) {
    size_t i = (size_t)blockIdx.x * 256 + threadIdx.x;
    if (i >= n) return;
    int bc = (int)(i >> 14);
    o[i] = x[i] * (1.f + g[bc]);
}

__global__ void k_rgb(const float* __restrict__ x, const float* __restrict__ Rw,
                      const float* __restrict__ rb, float* __restrict__ rgb) {
    size_t i = (size_t)blockIdx.x * 256 + threadIdx.x;     // B*P threads
    if (i >= (size_t)NB * HWP) return;
    int b = (int)(i >> 14);
    int p = (int)(i & (HWP - 1));
    float s = rb[0];
    for (int c = 0; c < NC; ++c)
        s += Rw[c] * x[((size_t)(b * NC + c) << 14) + p];
    rgb[i] = s;
}

__global__ void k_scale_sp(float* __restrict__ x, const float* __restrict__ sc, size_t n) {
    size_t i = (size_t)blockIdx.x * 256 + threadIdx.x;
    if (i >= n) return;
    int p = (int)(i & (HWP - 1));
    int b = (int)(i >> 20);
    x[i] *= (1.f + sc[((size_t)b << 14) + p]);
}

__global__ void k_dynacc(const float* __restrict__ kt, const float* __restrict__ src,
                         const float* __restrict__ base, float* __restrict__ out,
                         int dr, int dc, int init, size_t n) {
    size_t i = (size_t)blockIdx.x * 256 + threadIdx.x;
    if (i >= n) return;
    int w  = (int)(i & 127);
    int hh = (int)((i >> 7) & 127);
    int h2 = iclamp(hh + dr, 0, HH - 1);
    int w2 = iclamp(w + dc, 0, WW - 1);
    size_t j = (i & ~(size_t)(HWP - 1)) + (h2 << 7) + w2;
    float v = kt[i] * src[j];
    out[i] = init ? (base[i] + v) : (out[i] + v);
}

__global__ void k_maxmean(const float* __restrict__ sa, float* __restrict__ comp) {
    size_t i = (size_t)blockIdx.x * 256 + threadIdx.x;     // B*P threads
    if (i >= (size_t)NB * HWP) return;
    int b = (int)(i >> 14);
    int p = (int)(i & (HWP - 1));
    float mx = -3.4e38f, s = 0.f;
    for (int c = 0; c < NC; ++c) {
        float v = sa[((size_t)(b * NC + c) << 14) + p];
        mx = v > mx ? v : mx;
        s += v;
    }
    comp[((size_t)(b * 2 + 0) << 14) + p] = mx;
    comp[((size_t)(b * 2 + 1) << 14) + p] = s * (1.f / (float)NC);
}

__global__ void k_spatial(const float* __restrict__ comp, const float* __restrict__ Ew,
                          const float* __restrict__ eb, float* __restrict__ scl) {
    size_t i = (size_t)blockIdx.x * 256 + threadIdx.x;     // B*P threads
    if (i >= (size_t)NB * HWP) return;
    int b = (int)(i >> 14);
    int p = (int)(i & (HWP - 1));
    int h = p >> 7, w = p & 127;
    float s = eb[0];
    for (int c2 = 0; c2 < 2; ++c2)
        for (int di = 0; di < 5; ++di) {
            int hh = h + di - 2;
            if (hh < 0 || hh >= HH) continue;
            for (int dj = 0; dj < 5; ++dj) {
                int wv = w + dj - 2;
                if (wv < 0 || wv >= WW) continue;
                s += Ew[c2 * 25 + di * 5 + dj] *
                     comp[((size_t)(b * 2 + c2) << 14) + hh * WW + wv];
            }
        }
    scl[i] = 1.f / (1.f + expf(-s));
}

__global__ void k_fe(const float* __restrict__ x, const float* __restrict__ sc,
                     _Float16* __restrict__ o, size_t n) {
    size_t i = (size_t)blockIdx.x * 256 + threadIdx.x;
    if (i >= n) return;
    int p = (int)(i & (HWP - 1));
    int b = (int)(i >> 20);
    o[i] = (_Float16)(x[i] * (1.f + sc[((size_t)b << 14) + p]));
}

// ---------------------------------------------------------------------------
extern "C" void kernel_launch(void* const* d_in, const int* in_sizes, int n_in,
                              void* d_out, int out_size, void* d_ws, size_t ws_size,
                              hipStream_t stream) {
    const float* f_event = (const float*)d_in[0];
    const float* f_blur  = (const float*)d_in[1];
    const float* Wt1 = (const float*)d_in[2];
    const float* Wt2 = (const float*)d_in[3];
    const float* Wt3 = (const float*)d_in[4];
    const float* A1  = (const float*)d_in[5];
    const float* b1  = (const float*)d_in[6];
    const float* A2  = (const float*)d_in[7];
    const float* b2  = (const float*)d_in[8];
    const float* G1  = (const float*)d_in[9];
    const float* g1b = (const float*)d_in[10];
    const float* G2  = (const float*)d_in[11];
    const float* g2b = (const float*)d_in[12];
    const float* Rw  = (const float*)d_in[13];
    const float* rb  = (const float*)d_in[14];
    const float* S1  = (const float*)d_in[15];
    const float* S2  = (const float*)d_in[16];
    const float* Ew  = (const float*)d_in[17];
    const float* eb  = (const float*)d_in[18];
    const float* Fw  = (const float*)d_in[19];

    const size_t E  = EELEM;
    const size_t BP = (size_t)NB * HWP;

    char* ws = (char*)d_ws;
    _Float16* hA  = (_Float16*)ws; ws += E * 2;
    _Float16* hB  = (_Float16*)ws; ws += E * 2;
    _Float16* hEv = (_Float16*)ws; ws += E * 2;
    _Float16* hG  = (_Float16*)ws; ws += E * 2;
    float* F1 = (float*)ws; ws += E * 4;     // fused32, later sa32
    float* F2 = (float*)ws; ws += E * 4;     // fused2, later fb_dyn
    float* F3 = (float*)ws; ws += E * 4;     // fb32
    float* KT = (float*)ws; ws += E * 4;     // per-tap dynamic kernel
    float* rgb  = (float*)ws; ws += BP * 4;
    float* comp = (float*)ws; ws += 2 * BP * 4;
    float* scl  = (float*)ws; ws += BP * 4;
    float* w0   = (float*)ws; ws += 512 * 4;
    float* w0s  = (float*)ws; ws += 512 * 4;
    _Float16* Wt1p = (_Float16*)ws; ws += 36864 * 2;
    _Float16* Wt2p = (_Float16*)ws; ws += 36864 * 2;
    _Float16* Wt3p = (_Float16*)ws; ws += 36864 * 2;
    _Float16* G1p  = (_Float16*)ws; ws += 36864 * 2;
    _Float16* S1p  = (_Float16*)ws; ws += 36864 * 2;
    _Float16* S2p  = (_Float16*)ws; ws += 36864 * 2;
    _Float16* G2p  = (_Float16*)ws; ws += 36864 * 2;
    _Float16* Fwp  = (_Float16*)ws; ws += 16384 * 2;

    const dim3 blk(32, 8);
    const dim3 g3(HWP / 32, NB);      // Cout=64 convs, NPIX=32
    const dim3 g1f(HWP / 16, NB);     // Cout=128 fusion conv, NPIX=16
    const int EB = (int)(E / 256);
    const int PB = (int)((BP + 255) / 256);
    const int WB = (36864 + 255) / 256;

    // --- weight packing (f32 -> f16, tap-major) ---
    k_pack3<<<WB, 256, 0, stream>>>(Wt1, Wt1p, 64, 64);
    k_pack3<<<WB, 256, 0, stream>>>(Wt2, Wt2p, 64, 64);
    k_pack3<<<WB, 256, 0, stream>>>(Wt3, Wt3p, 64, 64);
    k_pack3<<<WB, 256, 0, stream>>>(G1,  G1p,  64, 64);
    k_pack3<<<WB, 256, 0, stream>>>(S1,  S1p,  64, 64);
    k_pack3<<<WB, 256, 0, stream>>>(S2,  S2p,  64, 64);
    k_packg2<<<WB, 256, 0, stream>>>(G2, G2p);
    k_cast_h<<<64, 256, 0, stream>>>(Fw, Fwp, 16384);

    // --- activations to f16 ---
    k_add2h<<<EB, 256, 0, stream>>>(f_event, f_blur, hA, E);   // fused0
    k_cast_h<<<EB, 256, 0, stream>>>(f_event, hEv, E);

    // --- trans: conv3x3 -> relu -> conv3x3 ---
    k_conv_wmma<3,64,64,32><<<g3, blk, 0, stream>>>(Wt1p, hA, hA, 64, nullptr, hB,
                                                    nullptr, nullptr, 0, 0, 1);
    k_conv_wmma<3,64,64,32><<<g3, blk, 0, stream>>>(Wt2p, hB, hB, 64, F1, hA,
                                                    nullptr, nullptr, 0, 0, 0);

    // --- SE channel gate ---
    k_cmean<<<NB * NC, 256, 0, stream>>>(F1, w0);
    k_mlp<<<1, 256, 0, stream>>>(w0, A1, b1, A2, b2, w0s);
    k_gate_ch<<<EB, 256, 0, stream>>>(f_blur, w0s, F3, E);     // f_b

    // --- trans2 + residual ---
    k_conv_wmma<3,64,64,32><<<g3, blk, 0, stream>>>(Wt3p, hA, hA, 64, F2, nullptr,
                                                    F1, nullptr, 0, 0, 1);

    // --- rgb attention gate on f_b ---
    k_rgb<<<PB, 256, 0, stream>>>(F2, Rw, rb, rgb);
    k_scale_sp<<<EB, 256, 0, stream>>>(F3, rgb, E);

    // --- dynamic per-pixel kernel: G1 conv, then 9 per-tap GEMMs + apply ---
    k_conv_wmma<3,64,64,32><<<g3, blk, 0, stream>>>(G1p, hEv, hEv, 64, nullptr, hG,
                                                    nullptr, g1b, 1, 0, 1);
    for (int tap = 0; tap < 9; ++tap) {
        k_conv_wmma<1,64,64,32><<<g3, blk, 0, stream>>>(G2p + (size_t)tap * 4096,
                                                        hG, hG, 64, KT, nullptr,
                                                        nullptr, g2b, 9, tap, 0);
        k_dynacc<<<EB, 256, 0, stream>>>(KT, F3, F3, F2,
                                         tap / 3 - 1, tap % 3 - 1,
                                         tap == 0 ? 1 : 0, E);
    }

    // --- spatial attention on events ---
    k_conv_wmma<3,64,64,32><<<g3, blk, 0, stream>>>(S1p, hEv, hEv, 64, nullptr, hB,
                                                    nullptr, nullptr, 0, 0, 1);
    k_conv_wmma<3,64,64,32><<<g3, blk, 0, stream>>>(S2p, hB, hB, 64, F1, nullptr,
                                                    nullptr, nullptr, 0, 0, 0);
    k_maxmean<<<PB, 256, 0, stream>>>(F1, comp);
    k_spatial<<<PB, 256, 0, stream>>>(comp, Ew, eb, scl);
    k_fe<<<EB, 256, 0, stream>>>(f_event, scl, hA, E);         // f_e (f16)
    k_cast_h<<<EB, 256, 0, stream>>>(F2, hG, E);               // f_b final (f16)

    // --- fusion 1x1 conv on concat(f_e, f_b): 128 -> 128, f32 out ---
    k_conv_wmma<1,128,128,16><<<g1f, blk, 0, stream>>>(Fwp, hA, hG, 64,
                                                       (float*)d_out, nullptr,
                                                       nullptr, nullptr, 0, 0, 0);
}
